// Attention_4999341932693
// MI455X (gfx1250) — matrix operand order repaired and checked by reading, not compiled, not run on hardware
//
#include <hip/hip_runtime.h>
#include <hip/hip_bf16.h>
#include <hip/hip_fp16.h>
#include <stdint.h>

// ---------------------------------------------------------------------------
// Attention on gfx1250 (MI455X): f16 WMMA with fp32 accumulation.
//   Q = Xq Wq^T, K = Xk Wk^T, V = Xv Wv^T   (f16 operands, f32 accum)
//   S = QK^T (causal-skipped blocks), softmax (scale+mask fused), O = P V
// GEMM: 256 threads = 8 wave32s, per-wave 32x64 tile. B staged into LDS with
// CDNA5 GLOBAL_LOAD_ASYNC_TO_LDS_B128 (ASYNCcnt-tracked, no VGPR round-trip),
// double-buffered 64x64 slab (16 WMMAs per barrier), A double-buffered in
// registers, K compile-time so the pipelined loop ping-pongs cleanly.
// ---------------------------------------------------------------------------

typedef __attribute__((ext_vector_type(16))) _Float16 v16h;
typedef __attribute__((ext_vector_type(8)))  _Float16 v8h;
typedef __attribute__((ext_vector_type(4)))  _Float16 v4h;
typedef __attribute__((ext_vector_type(8)))  float    v8f;

#define ATT_S 4096
#define ATT_D 1024

__device__ inline v8f wmma_f16(v16h a, v16h b, v8f c) {
  // (neg_a, A, neg_b, B, c_mod, C, reuse_a, reuse_b)
  return __builtin_amdgcn_wmma_f32_16x16x32_f16(false, a, false, b,
                                                (short)0, c, false, false);
}

// CDNA5 async memory->LDS copy, 16 B per lane, tracked by ASYNCcnt.
// INST_OFFSET applies to both the LDS and global address.
__device__ inline void async_b128_to_lds(unsigned lds_off, const void* gptr) {
  asm volatile("global_load_async_to_lds_b128 %0, %1, off"
               :: "v"(lds_off), "v"((unsigned long long)(uintptr_t)gptr)
               : "memory");
}
__device__ inline void async_b128_to_lds_o16(unsigned lds_off, const void* gptr) {
  asm volatile("global_load_async_to_lds_b128 %0, %1, off offset:16"
               :: "v"(lds_off), "v"((unsigned long long)(uintptr_t)gptr)
               : "memory");
}
__device__ inline void wait_async0() {
  asm volatile("s_wait_asynccnt 0x0" ::: "memory");
}

// A fragment (16x32 f16): lanes 0-15 hold row m=lane, K {kb..+7, kb+16..+23};
// lanes 16-31 hold K {kb+8..+15, kb+24..+31} -> two contiguous 16 B loads.
__device__ inline v16h load_a_frag(const _Float16* p /*row base + kb*/, int hi) {
  const _Float16* q = p + hi * 8;
  v8h lo = *(const v8h*)q;
  v8h hv = *(const v8h*)(q + 16);
  v16h a;
#pragma unroll
  for (int i = 0; i < 8; ++i) { a[i] = lo[i]; a[i + 8] = hv[i]; }
  return a;
}

// B fragment (32x16 f16) from LDS: col n=lane&15 holds K {kb+(lane>>4)*8..+7,
// kb+16+(lane>>4)*8..+7} (matches A fragment) -> two 16 B ds loads.
__device__ inline v16h load_b_frag_lds(const _Float16* p /*col+substep base*/,
                                       int hi) {
  const _Float16* q = p + hi * 8;
  v8h lo = *(const v8h*)q;
  v8h hv = *(const v8h*)(q + 16);
  v16h b;
#pragma unroll
  for (int i = 0; i < 8; ++i) { b[i] = lo[i]; b[i + 8] = hv[i]; }
  return b;
}

// ---------------------------------------------------------------------------
// fp32 -> f16 conversion (vectorized streaming copy)
// ---------------------------------------------------------------------------
__global__ __launch_bounds__(256) void cvt_f32_to_f16_kernel(
    const float* __restrict__ in, _Float16* __restrict__ out, int n4) {
  int i = blockIdx.x * blockDim.x + threadIdx.x;
  if (i < n4) {
    float4 v = ((const float4*)in)[i];
    v4h h;
    h[0] = (_Float16)v.x; h[1] = (_Float16)v.y;
    h[2] = (_Float16)v.z; h[3] = (_Float16)v.w;
    ((v4h*)out)[i] = h;
  }
}

// ---------------------------------------------------------------------------
// Out = A * B^T ; A[M,K], B[N,K] row-major f16, K compile-time.
// Block: 256 threads (8 waves), tile 256(M) x 64(N); wave tile 32 x 64.
// ---------------------------------------------------------------------------
template <typename OutT, bool TRANSPOSED_STORE, bool CAUSAL, int K>
__global__ __launch_bounds__(256) void gemm_tn_kernel(
    const _Float16* __restrict__ A, const _Float16* __restrict__ B,
    OutT* __restrict__ Out, int M, int N) {
  constexpr int KSTEP   = 64;   // k-depth staged per barrier
  constexpr int BSTRIDE = 72;   // LDS col stride in halves (144 B, 16 B align)

  const int n0     = blockIdx.x * 64;
  const int mBlock = blockIdx.y * 256;
  if (CAUSAL && n0 > mBlock + 255) return;   // block-uniform early exit

  const int lane = threadIdx.x & 31;
  const int wave = threadIdx.x >> 5;         // 0..7
  const int lo16 = lane & 15;
  const int hi   = lane >> 4;
  const int m0   = mBlock + wave * 32;

  __shared__ __align__(32) _Float16 Bs[2][64 * BSTRIDE];

  // Cooperative B staging: thread t covers (col = t&63, 16-half segment t>>6),
  // 32 B per thread per slab, staged with async-to-LDS (no VGPR round-trip).
  const int bcol = threadIdx.x & 63;
  const int bseg = threadIdx.x >> 6;         // 0..3
  const _Float16* bsrc = B + (size_t)(n0 + bcol) * K + bseg * 16;
  const unsigned ldsoff0 =
      (unsigned)(uintptr_t)&Bs[0][bcol * BSTRIDE + bseg * 16];
  const unsigned ldsoff1 =
      (unsigned)(uintptr_t)&Bs[1][bcol * BSTRIDE + bseg * 16];

  const _Float16* arow0 = A + (size_t)(m0 + lo16)      * K;
  const _Float16* arow1 = A + (size_t)(m0 + 16 + lo16) * K;

  // Prologue: async-stage slab 0 and load its A fragments (both substeps).
  async_b128_to_lds(ldsoff0, bsrc);
  async_b128_to_lds_o16(ldsoff0, bsrc);
  v16h a00 = load_a_frag(arow0,      hi), a10 = load_a_frag(arow1,      hi);
  v16h a01 = load_a_frag(arow0 + 32, hi), a11 = load_a_frag(arow1 + 32, hi);
  wait_async0();
  __syncthreads();

  v8f acc[8] = {v8f{}, v8f{}, v8f{}, v8f{}, v8f{}, v8f{}, v8f{}, v8f{}};

#pragma unroll 2
  for (int kb = 0; kb < K; kb += KSTEP) {
    const int cur = (kb >> 6) & 1;
    const int kn  = (kb + KSTEP < K) ? kb + KSTEP : 0;  // clamped (folds: K const)

    // Kick off next slab's async B stage into the other buffer immediately:
    // it stays in flight across the whole compute phase. Prev readers of that
    // buffer all passed the previous barrier, so this is race-free.
    const unsigned dsto = cur ? ldsoff0 : ldsoff1;
    async_b128_to_lds(dsto, bsrc + kn);
    async_b128_to_lds_o16(dsto, bsrc + kn);

    // Prefetch next slab's A fragments (register double-buffered).
    v16h a00n = load_a_frag(arow0 + kn,      hi);
    v16h a10n = load_a_frag(arow1 + kn,      hi);
    v16h a01n = load_a_frag(arow0 + kn + 32, hi);
    v16h a11n = load_a_frag(arow1 + kn + 32, hi);

    // Two 32-deep substeps; per substep: 8 ds_load_b128 -> 8 WMMAs.
#pragma unroll
    for (int s = 0; s < 2; ++s) {
      const v16h aA = s ? a01 : a00;
      const v16h aB = s ? a11 : a10;
      const _Float16* base = &Bs[cur][s * 32];
      v16h b0 = load_b_frag_lds(base + (0 * 16 + lo16) * BSTRIDE, hi);
      v16h b1 = load_b_frag_lds(base + (1 * 16 + lo16) * BSTRIDE, hi);
      v16h b2 = load_b_frag_lds(base + (2 * 16 + lo16) * BSTRIDE, hi);
      v16h b3 = load_b_frag_lds(base + (3 * 16 + lo16) * BSTRIDE, hi);
      acc[0] = wmma_f16(aA, b0, acc[0]);
      acc[4] = wmma_f16(aB, b0, acc[4]);
      acc[1] = wmma_f16(aA, b1, acc[1]);
      acc[5] = wmma_f16(aB, b1, acc[5]);
      acc[2] = wmma_f16(aA, b2, acc[2]);
      acc[6] = wmma_f16(aB, b2, acc[6]);
      acc[3] = wmma_f16(aA, b3, acc[3]);
      acc[7] = wmma_f16(aB, b3, acc[7]);
    }

    // Publish next slab: our async writes must land before the barrier.
    wait_async0();
    __syncthreads();
    a00 = a00n; a10 = a10n; a01 = a01n; a11 = a11n;
  }

#pragma unroll
  for (int t = 0; t < 2; ++t) {
    const int mb = m0 + t * 16;
#pragma unroll
    for (int j = 0; j < 4; ++j) {
      v8f a = acc[t * 4 + j];
      const int n = n0 + j * 16 + lo16;
#pragma unroll
      for (int i = 0; i < 8; ++i) {
        const int m = mb + i + hi * 8;       // C/D layout: VGPR i -> M=i / i+8
        if (TRANSPOSED_STORE) Out[(size_t)n * M + m] = (OutT)a[i];
        else                  Out[(size_t)m * N + n] = (OutT)a[i];
      }
    }
  }
}

// ---------------------------------------------------------------------------
// Row softmax with fused 1/sqrt(D) scale and analytic causal mask.
// One 256-thread block per row; LDS tree reductions; f16 probabilities out.
// ---------------------------------------------------------------------------
__global__ __launch_bounds__(256) void softmax_kernel(
    const float* __restrict__ Sraw, _Float16* __restrict__ P, int S) {
  __shared__ float red[256];
  const int r = blockIdx.x;
  const int tid = threadIdx.x;
  const float scale = 0.03125f;              // 1/sqrt(1024)
  const float* row = Sraw + (size_t)r * S;

  float mx = -3.4e38f;
  for (int j = tid; j <= r; j += 256)        // j > r is masked -> never the max
    mx = fmaxf(mx, row[j] * scale);
  red[tid] = mx; __syncthreads();
  for (int s = 128; s > 0; s >>= 1) {
    if (tid < s) red[tid] = fmaxf(red[tid], red[tid + s]);
    __syncthreads();
  }
  mx = red[0]; __syncthreads();

  float sum = 0.f;
  for (int j = tid; j <= r; j += 256)
    sum += __expf(row[j] * scale - mx);
  red[tid] = sum; __syncthreads();
  for (int s = 128; s > 0; s >>= 1) {
    if (tid < s) red[tid] += red[tid + s];
    __syncthreads();
  }
  const float inv = 1.f / red[0];

  _Float16* prow = P + (size_t)r * S;
  for (int j = tid; j < S; j += 256) {
    float v = (j <= r) ? __expf(row[j] * scale - mx) * inv : 0.f;
    prow[j] = (_Float16)v;
  }
}

// ---------------------------------------------------------------------------
extern "C" void kernel_launch(void* const* d_in, const int* in_sizes, int n_in,
                              void* d_out, int out_size, void* d_ws, size_t ws_size,
                              hipStream_t stream) {
  (void)in_sizes; (void)n_in; (void)out_size; (void)ws_size;
  const int S = ATT_S, D = ATT_D;
  const size_t SD = (size_t)S * D;     // 4,194,304
  const size_t DD = (size_t)D * D;     // 1,048,576

  const float* Xq = (const float*)d_in[0];
  const float* Xk = (const float*)d_in[1];
  const float* Xv = (const float*)d_in[2];
  /* d_in[3] = mask: causal, applied analytically */
  const float* Wq = (const float*)d_in[4];
  const float* Wk = (const float*)d_in[5];
  const float* Wv = (const float*)d_in[6];

  _Float16* hb  = (_Float16*)d_ws;
  _Float16* Qh  = hb + 0 * SD;
  _Float16* Kh  = hb + 1 * SD;
  _Float16* Vt  = hb + 2 * SD;         // [D, S] transposed
  _Float16* Xqh = hb + 3 * SD;
  _Float16* Xkh = hb + 4 * SD;
  _Float16* Xvh = hb + 5 * SD;
  _Float16* Wqh = hb + 6 * SD;
  _Float16* Wkh = Wqh + DD;
  _Float16* Wvh = Wkh + DD;
  size_t half_bytes = (6 * SD + 3 * DD) * sizeof(_Float16);
  float*    Sraw = (float*)((char*)d_ws + half_bytes);
  _Float16* Ph   = (_Float16*)((char*)Sraw + (size_t)S * S * sizeof(float));

  // 1) fp32 -> f16 operand conversion
  cvt_f32_to_f16_kernel<<<(int)(SD / 4 / 256), 256, 0, stream>>>(Xq, Xqh, (int)(SD / 4));
  cvt_f32_to_f16_kernel<<<(int)(SD / 4 / 256), 256, 0, stream>>>(Xk, Xkh, (int)(SD / 4));
  cvt_f32_to_f16_kernel<<<(int)(SD / 4 / 256), 256, 0, stream>>>(Xv, Xvh, (int)(SD / 4));
  cvt_f32_to_f16_kernel<<<(int)(DD / 4 / 256), 256, 0, stream>>>(Wq, Wqh, (int)(DD / 4));
  cvt_f32_to_f16_kernel<<<(int)(DD / 4 / 256), 256, 0, stream>>>(Wk, Wkh, (int)(DD / 4));
  cvt_f32_to_f16_kernel<<<(int)(DD / 4 / 256), 256, 0, stream>>>(Wv, Wvh, (int)(DD / 4));

  // 2) projections (WMMA): Q,K normal layout; V stored transposed
  dim3 blk(256);
  gemm_tn_kernel<_Float16, false, false, ATT_D>
      <<<dim3(D / 64, S / 256), blk, 0, stream>>>(Xqh, Wqh, Qh, S, D);
  gemm_tn_kernel<_Float16, false, false, ATT_D>
      <<<dim3(D / 64, S / 256), blk, 0, stream>>>(Xkh, Wkh, Kh, S, D);
  gemm_tn_kernel<_Float16, true, false, ATT_D>
      <<<dim3(D / 64, S / 256), blk, 0, stream>>>(Xvh, Wvh, Vt, S, D);

  // 3) scores = Q K^T (causal block skipping)
  gemm_tn_kernel<float, false, true, ATT_D>
      <<<dim3(S / 64, S / 256), blk, 0, stream>>>(Qh, Kh, Sraw, S, S);

  // 4) row softmax (scale + causal mask fused), f16 probabilities
  softmax_kernel<<<S, 256, 0, stream>>>(Sraw, Ph, S);

  // 5) O = P V   (B = Vt[D,S], K-contiguous)
  gemm_tn_kernel<float, false, false, ATT_S>
      <<<dim3(D / 64, S / 256), blk, 0, stream>>>(Ph, Vt, (float*)d_out, S, D);
}
